// MultiHeadAttention_90726889161019
// MI455X (gfx1250) — matrix operand order repaired and checked by reading, not compiled, not run on hardware
//
#include <hip/hip_runtime.h>

// MI455X / gfx1250: wave32, WMMA 16x16x32 bf16, f32 accumulate.
// Flash-attention forward: qkv proj -> online-softmax attention -> out proj.

#define DEV_INLINE __device__ __forceinline__

constexpr int kH  = 16;    // heads
constexpr int kHS = 64;    // head size
constexpr int kC  = 1024;  // channels
constexpr int kB  = 2;     // batch
constexpr int kT  = 2048;  // seq len
constexpr int kHH = kH * kHS; // 1024

typedef __attribute__((ext_vector_type(16))) __bf16        v16bf;
typedef __attribute__((ext_vector_type(8)))  float         v8f;
typedef __attribute__((ext_vector_type(4)))  unsigned int  v4u;

union Frag16 { v16bf bf; v4u q[2]; };

// A-matrix fragment, 16(M) x 32(K) bf16, src row-major with leading dim `ld`.
// ISA layout: lanes 0-15 hold M=lane, K=0..7 & 16..23; lanes 16-31 hold K=8..15 & 24..31.
DEV_INLINE v16bf load_a_frag(const __bf16* src, int ld, int lane) {
  const int hf = (lane >> 4) & 1;
  const int m  = lane & 15;
  const __bf16* p = src + (long)m * ld + 8 * hf;
  Frag16 f;
  f.q[0] = *(const v4u*)(p);
  f.q[1] = *(const v4u*)(p + 16);
  return f.bf;
}

// B-matrix fragment, 32(K) x 16(N) bf16. srcT is the TRANSPOSED operand:
// N x K row-major, so each lane (N = lane&15) reads two 8-wide K runs
// (lanes 0-15: K=0..7 & 16..23, lanes 16-31: K=8..15 & 24..31).
DEV_INLINE v16bf load_b_frag(const __bf16* srcT, int ld, int lane) {
  const int kh = (lane >> 4) & 1;
  const int n  = lane & 15;
  const __bf16* p = srcT + (long)n * ld + 8 * kh;
  Frag16 f;
  f.q[0] = *(const v4u*)(p);
  f.q[1] = *(const v4u*)(p + 16);
  return f.bf;
}

DEV_INLINE v8f wmma_bf16(v16bf a, v16bf b, v8f c) {
  return __builtin_amdgcn_wmma_f32_16x16x32_bf16(
      /*neg_a=*/false, a, /*neg_b=*/false, b,
      /*c_mod=*/(short)0, c, /*reuse_a=*/false, /*reuse_b=*/false);
}

// ---------------- conversion / transpose kernels ----------------

__global__ void __launch_bounds__(256) k_cvt_bf16(const float* __restrict__ in,
                                                  __bf16* __restrict__ out, long n) {
  long i = (long)blockIdx.x * blockDim.x + threadIdx.x;
  long stride = (long)gridDim.x * blockDim.x;
  for (; i < n; i += stride) out[i] = (__bf16)in[i];
}

// [H, C, HS] f32 -> [H, HS, C] bf16 (per-head transpose for B-fragments)
__global__ void __launch_bounds__(256) k_twh(const float* __restrict__ in,
                                             __bf16* __restrict__ out) {
  const long n = (long)kH * kC * kHS;
  long i = (long)blockIdx.x * blockDim.x + threadIdx.x;
  long stride = (long)gridDim.x * blockDim.x;
  for (; i < n; i += stride) {
    int h = (int)(i / (kC * kHS));
    int r = (int)(i % (kC * kHS));
    int c = r / kHS;
    int d = r % kHS;
    out[((long)h * kHS + d) * kC + c] = (__bf16)in[i];
  }
}

// Wp [H*HS, C] f32 -> WpT [C, H*HS] bf16
__global__ void __launch_bounds__(256) k_twp(const float* __restrict__ in,
                                             __bf16* __restrict__ out) {
  const long n = (long)kHH * kC;
  long i = (long)blockIdx.x * blockDim.x + threadIdx.x;
  long stride = (long)gridDim.x * blockDim.x;
  for (; i < n; i += stride) {
    int k = (int)(i / kC);
    int c = (int)(i % kC);
    out[(long)c * kHH + k] = (__bf16)in[i];
  }
}

// ---------------- QKV projection ----------------
// One wave per (b, h, 16-row tile of T). q,k stored [B,H,T,HS]; v stored
// transposed [B,H,HS,T] so the P@V B-fragment is contiguous. Scale 1/sqrt(HS)
// folded into q.
__global__ void __launch_bounds__(256) k_qkv(const __bf16* __restrict__ Xb,
                                             const __bf16* __restrict__ WqT,
                                             const __bf16* __restrict__ WkT,
                                             const __bf16* __restrict__ WvT,
                                             __bf16* __restrict__ qbp,
                                             __bf16* __restrict__ kbp,
                                             __bf16* __restrict__ vTp) {
  const int lane = threadIdx.x & 31;
  const int wid  = blockIdx.x * (blockDim.x >> 5) + (threadIdx.x >> 5);
  const int TT = kT / 16;
  if (wid >= kB * kH * TT) return;
  const int tt = wid % TT;
  const int h  = (wid / TT) % kH;
  const int b  = wid / (TT * kH);
  const int t0 = tt * 16;
  const long bh = (long)b * kH + h;

  const __bf16* xrow = Xb + ((long)b * kT + t0) * kC;
  const int hf = lane >> 4, nl = lane & 15;

  for (int m = 0; m < 3; ++m) {
    const __bf16* W = (m == 0) ? (WqT + (long)h * kHS * kC)
                    : (m == 1) ? (WkT + (long)h * kHS * kC)
                               : (WvT + (long)h * kHS * kC);
    v8f acc[4] = {};
    for (int kc = 0; kc < kC / 32; ++kc) {
      v16bf a = load_a_frag(xrow + kc * 32, kC, lane);
#pragma unroll
      for (int nt = 0; nt < 4; ++nt) {
        v16bf bf = load_b_frag(W + (long)(nt * 16) * kC + kc * 32, kC, lane);
        acc[nt] = wmma_bf16(a, bf, acc[nt]);
      }
    }
    if (m < 2) {
      const float scale = (m == 0) ? 0.125f : 1.0f;  // 1/sqrt(64) on q
      __bf16* dst = ((m == 0) ? qbp : kbp) + (bh * kT + t0) * kHS;
#pragma unroll
      for (int nt = 0; nt < 4; ++nt)
#pragma unroll
        for (int j = 0; j < 8; ++j)
          dst[(long)(j + 8 * hf) * kHS + nt * 16 + nl] = (__bf16)(acc[nt][j] * scale);
    } else {
      __bf16* dst = vTp + bh * kHS * kT + t0;
#pragma unroll
      for (int nt = 0; nt < 4; ++nt)
#pragma unroll
        for (int j = 0; j < 8; ++j)
          dst[(long)(nt * 16 + nl) * kT + (j + 8 * hf)] = (__bf16)acc[nt][j];
    }
  }
}

// ---------------- flash attention ----------------
// One wave per (b, h, 32-row query block = two 16-row WMMA tiles). The two
// q-tiles SHARE every K and V B-fragment load, doubling WMMA per loaded byte
// vs a 16-row wave and halving K/V traffic. Online softmax per tile;
// probabilities round-trip through LDS to convert the WMMA D-layout into an
// A-layout fragment for the P@V matmuls.
__global__ void __launch_bounds__(256) k_attn(const __bf16* __restrict__ qbp,
                                              const __bf16* __restrict__ kbp,
                                              const __bf16* __restrict__ vTp,
                                              __bf16* __restrict__ Ob) {
  __shared__ __align__(16) __bf16 Pshm[8][2][16][32];  // per-wave P tiles (16 KB)
  const int lane = threadIdx.x & 31;
  const int w    = threadIdx.x >> 5;
  const int wid  = blockIdx.x * (blockDim.x >> 5) + w;
  const int QT = kT / 32;  // 64 query blocks of 32 rows
  if (wid >= kB * kH * QT) return;
  const int qt = wid % QT;
  const int h  = (wid / QT) % kH;
  const int b  = wid / (QT * kH);
  const int t0 = qt * 32;
  const long bh = (long)b * kH + h;
  const int hf = lane >> 4, nl = lane & 15;

  const __bf16* qbase = qbp + (bh * kT + t0) * kHS;
  const __bf16* kbase = kbp + bh * kT * kHS;
  const __bf16* vbase = vTp + bh * kHS * kT;

  v16bf aq[2][2];
#pragma unroll
  for (int q2 = 0; q2 < 2; ++q2) {
    aq[q2][0] = load_a_frag(qbase + (long)q2 * 16 * kHS, kHS, lane);       // hs 0..31
    aq[q2][1] = load_a_frag(qbase + (long)q2 * 16 * kHS + 32, kHS, lane);  // hs 32..63
  }

  v8f o[2][4] = {};
  float mi[2][8], li[2][8];
#pragma unroll
  for (int q2 = 0; q2 < 2; ++q2)
#pragma unroll
    for (int j = 0; j < 8; ++j) { mi[q2][j] = -1e30f; li[q2][j] = 0.0f; }

  const int nblocks = t0 / 32 + 1;  // causal: keys 0 .. t0+31 (t0 % 32 == 0)
  for (int blk = 0; blk < nblocks; ++blk) {
    const int s0 = blk * 32;
    if (blk + 1 < nblocks) {  // pull next key block toward the WGP early
      __builtin_prefetch(kbase + (long)(s0 + 32) * kHS, 0, 1);
      __builtin_prefetch(vbase + s0 + 32, 0, 1);
    }
    // ---- shared K fragments for this 32-key block ----
    const v16bf bk00 = load_b_frag(kbase + (long)s0 * kHS, kHS, lane);
    const v16bf bk01 = load_b_frag(kbase + (long)s0 * kHS + 32, kHS, lane);
    const v16bf bk10 = load_b_frag(kbase + (long)(s0 + 16) * kHS, kHS, lane);
    const v16bf bk11 = load_b_frag(kbase + (long)(s0 + 16) * kHS + 32, kHS, lane);

#pragma unroll
    for (int q2 = 0; q2 < 2; ++q2) {
      // ---- S = q @ k^T (two 16x16 tiles over the 32 keys) ----
      v8f st0 = {}, st1 = {};
      st0 = wmma_bf16(aq[q2][0], bk00, st0);
      st0 = wmma_bf16(aq[q2][1], bk01, st0);
      st1 = wmma_bf16(aq[q2][0], bk10, st1);
      st1 = wmma_bf16(aq[q2][1], bk11, st1);
      // ---- causal mask on the diagonal block ----
      if (blk == nblocks - 1) {
#pragma unroll
        for (int j = 0; j < 8; ++j) {
          const int row = t0 + q2 * 16 + j + 8 * hf;
          st0[j] = ((s0 + nl)      > row) ? -1e30f : st0[j];
          st1[j] = ((s0 + 16 + nl) > row) ? -1e30f : st1[j];
        }
      }
      // ---- online softmax (row reductions across 16-lane halves) ----
      float newm[8], alpha[8];
#pragma unroll
      for (int j = 0; j < 8; ++j) {
        float v = fmaxf(st0[j], st1[j]);
        v = fmaxf(v, __shfl_xor(v, 1, 32));
        v = fmaxf(v, __shfl_xor(v, 2, 32));
        v = fmaxf(v, __shfl_xor(v, 4, 32));
        v = fmaxf(v, __shfl_xor(v, 8, 32));
        newm[j]     = fmaxf(mi[q2][j], v);
        alpha[j]    = __expf(mi[q2][j] - newm[j]);
        mi[q2][j]   = newm[j];
      }
#pragma unroll
      for (int j = 0; j < 8; ++j) {
        const float p0 = __expf(st0[j] - newm[j]);
        const float p1 = __expf(st1[j] - newm[j]);
        float s = p0 + p1;
        s += __shfl_xor(s, 1, 32);
        s += __shfl_xor(s, 2, 32);
        s += __shfl_xor(s, 4, 32);
        s += __shfl_xor(s, 8, 32);
        li[q2][j] = li[q2][j] * alpha[j] + s;
        o[q2][0][j] *= alpha[j]; o[q2][1][j] *= alpha[j];
        o[q2][2][j] *= alpha[j]; o[q2][3][j] *= alpha[j];
        // ---- D-layout -> A-layout via LDS ----
        Pshm[w][q2][j + 8 * hf][nl]      = (__bf16)p0;
        Pshm[w][q2][j + 8 * hf][16 + nl] = (__bf16)p1;
      }
    }
    asm volatile("s_wait_dscnt 0" ::: "memory");  // wave-local LDS RAW
    const v16bf ap0 = load_a_frag(&Pshm[w][0][0][0], 32, lane);
    const v16bf ap1 = load_a_frag(&Pshm[w][1][0][0], 32, lane);
    // ---- O += P @ V, V fragments shared by both q-tiles ----
#pragma unroll
    for (int nt = 0; nt < 4; ++nt) {
      const v16bf bv = load_b_frag(vbase + (long)(nt * 16) * kT + s0, kT, lane);
      o[0][nt] = wmma_bf16(ap0, bv, o[0][nt]);
      o[1][nt] = wmma_bf16(ap1, bv, o[1][nt]);
    }
  }

  // ---- normalize and store concat-head layout [B*T, H*HS] bf16 ----
#pragma unroll
  for (int q2 = 0; q2 < 2; ++q2) {
    __bf16* orow = Ob + ((long)b * kT + t0 + q2 * 16) * kHH + h * kHS;
#pragma unroll
    for (int j = 0; j < 8; ++j) {
      const float inv = 1.0f / li[q2][j];
      const long r = (long)(j + 8 * hf) * kHH;
      orow[r +  0 + nl] = (__bf16)(o[q2][0][j] * inv);
      orow[r + 16 + nl] = (__bf16)(o[q2][1][j] * inv);
      orow[r + 32 + nl] = (__bf16)(o[q2][2][j] * inv);
      orow[r + 48 + nl] = (__bf16)(o[q2][3][j] * inv);
    }
  }
}

// ---------------- output projection: out = Ob @ Wp + bp (f32 out) ----------------
__global__ void __launch_bounds__(256) k_oproj(const __bf16* __restrict__ Ob,
                                               const __bf16* __restrict__ WpT,
                                               const float* __restrict__ bp,
                                               float* __restrict__ out) {
  const int lane = threadIdx.x & 31;
  const int wid  = blockIdx.x * (blockDim.x >> 5) + (threadIdx.x >> 5);
  const int MT = (kB * kT) / 16, NT = kC / 16;  // 256 x 64 tiles
  if (wid >= MT * NT) return;
  const int nt = wid % NT, mt = wid / NT;

  const __bf16* arow = Ob  + (long)mt * 16 * kHH;
  const __bf16* brow = WpT + (long)nt * 16 * kHH;
  v8f acc = {};
  for (int kc = 0; kc < kHH / 32; ++kc) {
    v16bf a = load_a_frag(arow + kc * 32, kHH, lane);
    v16bf b = load_b_frag(brow + kc * 32, kHH, lane);
    acc = wmma_bf16(a, b, acc);
  }
  const int hf = lane >> 4, nl = lane & 15;
  const float bias = bp[nt * 16 + nl];
  float* orow = out + (long)mt * 16 * kC + nt * 16;
#pragma unroll
  for (int j = 0; j < 8; ++j)
    orow[(long)(j + 8 * hf) * kC + nl] = acc[j] + bias;
}

// ---------------- host launch ----------------

extern "C" void kernel_launch(void* const* d_in, const int* in_sizes, int n_in,
                              void* d_out, int out_size, void* d_ws, size_t ws_size,
                              hipStream_t stream) {
  const float* x  = (const float*)d_in[0];
  const float* Wq = (const float*)d_in[1];
  const float* Wk = (const float*)d_in[2];
  const float* Wv = (const float*)d_in[3];
  const float* Wp = (const float*)d_in[4];
  const float* bp = (const float*)d_in[5];
  float* out = (float*)d_out;

  // workspace carve-up (bf16, 256B aligned slabs); total ~48 MB
  size_t off = 0;
  auto alloc = [&](size_t bytes) {
    void* r = (char*)d_ws + off;
    off += (bytes + 255) & ~(size_t)255;
    return r;
  };
  __bf16* Xb  = (__bf16*)alloc((size_t)kB * kT * kC * 2);
  __bf16* WqT = (__bf16*)alloc((size_t)kH * kHS * kC * 2);
  __bf16* WkT = (__bf16*)alloc((size_t)kH * kHS * kC * 2);
  __bf16* WvT = (__bf16*)alloc((size_t)kH * kHS * kC * 2);
  __bf16* WpT = (__bf16*)alloc((size_t)kC * kHH * 2);
  __bf16* qb  = (__bf16*)alloc((size_t)kB * kH * kT * kHS * 2);
  __bf16* kb  = (__bf16*)alloc((size_t)kB * kH * kT * kHS * 2);
  __bf16* vT  = (__bf16*)alloc((size_t)kB * kH * kHS * kT * 2);
  __bf16* Ob  = (__bf16*)alloc((size_t)kB * kT * kHH * 2);

  k_cvt_bf16<<<2048, 256, 0, stream>>>(x, Xb, (long)kB * kT * kC);
  k_twh<<<1024, 256, 0, stream>>>(Wq, WqT);
  k_twh<<<1024, 256, 0, stream>>>(Wk, WkT);
  k_twh<<<1024, 256, 0, stream>>>(Wv, WvT);
  k_twp<<<1024, 256, 0, stream>>>(Wp, WpT);

  // 2*16*128 = 4096 waves, 8 waves per 256-thread block
  k_qkv<<<512, 256, 0, stream>>>(Xb, WqT, WkT, WvT, qb, kb, vT);
  // 2*16*64 = 2048 waves, 32 query rows each
  k_attn<<<256, 256, 0, stream>>>(qb, kb, vT, Ob);
  // 256*64 = 16384 tile-waves
  k_oproj<<<2048, 256, 0, stream>>>(Ob, WpT, bp, out);
}
